// EEGM2_S4_1812476199774
// MI455X (gfx1250) — compile-verified, matrix operand order checked by reading, not run on hardware
//
#include <hip/hip_runtime.h>
#include <cstdint>
#include <cstddef>

// ---------------------------------------------------------------------------
// Types / bf16 helpers (CDNA5 WMMA wants 16-lane __bf16 vectors, wave32)
// ---------------------------------------------------------------------------
typedef __attribute__((ext_vector_type(16))) unsigned short v16us;
typedef __attribute__((ext_vector_type(16))) __bf16         v16bf;
typedef __attribute__((ext_vector_type(2)))  __bf16         v2bf;
typedef __attribute__((ext_vector_type(8)))  float          v8f;
typedef __attribute__((ext_vector_type(8)))  unsigned       v8u;
typedef __attribute__((address_space(3)))    float          as3_float;

// Packed f32x2 -> bf16x2 (lowers to v_cvt_pk_bf16_f32)
__device__ __forceinline__ unsigned pack_bf16x2(float x, float y) {
  v2bf r;
  r[0] = (__bf16)x;
  r[1] = (__bf16)y;
  return __builtin_bit_cast(unsigned, r);
}

__device__ __forceinline__ v8f wmma_bf16(v16us a, v16us b, v8f c) {
  // D = A(16x32 bf16) * B(32x16 bf16) + C(16x16 f32)
  return __builtin_amdgcn_wmma_f32_16x16x32_bf16(
      false, __builtin_bit_cast(v16bf, a),
      false, __builtin_bit_cast(v16bf, b),
      (short)0, c, false, false);
}

// K-pair base per the ISA 16-bit 16x32 fragment layout (VGPR v, lane half)
__device__ __forceinline__ int kpair_base(int v, int half) {
  return ((v < 4) ? 2 * v : 16 + 2 * (v - 4)) + half * 8;
}

// CDNA5 async global->LDS copy, 16B per lane (GLOBAL_LOAD_ASYNC_TO_LDS_B128)
__device__ __forceinline__ void async_b128_to_lds(unsigned lds_addr,
                                                  unsigned long long gaddr) {
  asm volatile("global_load_async_to_lds_b128 %0, %1, off"
               :: "v"(lds_addr), "v"(gaddr)
               : "memory");
}
__device__ __forceinline__ void wait_async0() {
  asm volatile("s_wait_asynccnt 0x0" ::: "memory");
}

// ---------------------------------------------------------------------------
// WMMA GEMM:  C = act(A) * W[N,K]^T (+ bias)
//   AMODE==0          : A dense row-major (M,K)
//   AMODE==1, KW==1   : A read from channel-major (B,C1[+C2],L): pointwise
//   AMODE==1, KW==3   : implicit im2col conv1d, channels concat from A1|A2
//   CMODE==0          : C stored row-major (M,N)
//   CMODE==1          : C stored channel-major (B,N,L)
// Block: 128 threads = 4 waves; wave w owns rows [blockM+16w,+16) x 64 cols.
// The 64(K)x64(N) f32 weight tile is staged per block via async DMA to LDS
// (ASYNCcnt), overlapped with the A-fragment global loads; each thread then
// converts the region it DMA'd into a packed-bf16 tile (own-wave dependency
// only), so the WMMA fragment path reads raw u32 pairs from LDS.
// K must be a multiple of 64 (true at every call site). L = 1<<lSh.
// ---------------------------------------------------------------------------
template <int AMODE, int KW, int CMODE>
__global__ __launch_bounds__(128) void wmma_gemm_t(
    const float* __restrict__ A1, const float* __restrict__ A2,
    const float* __restrict__ W,  const float* __restrict__ bias,
    float* __restrict__ Cmat,
    int M, int N, int K,
    int C1, int C2, int Lin, int lSh, int pad)
{
  __shared__ float          Bsf[64][68];   // DMA target, f32; rows 16B-aligned
  __shared__ unsigned short Bs16[64][66];  // packed bf16 [n][k]; rows 4B-aligned

  const int tid  = threadIdx.x;
  const int lane = tid & 31;
  const int wave = tid >> 5;
  const int half = lane >> 4;
  const int lm   = lane & 15;
  const int tileM  = blockIdx.y * 64 + wave * 16;
  const int tileN0 = blockIdx.x * 64;

  const int row = tileM + lm;          // M is a multiple of 64 at all calls
  int cb = 0, cl = 0;
  if (AMODE == 1) { cb = row >> lSh; cl = row - (cb << lSh); }

  // B staging: thread copies column (tileN0+nload, clamped), k-range 32*khalf..
  const int nload = tid & 63;
  const int khalf = tid >> 6;                    // 0 or 1
  int ncl = tileN0 + nload;
  if (ncl >= N) ncl = N - 1;   // OOB columns computed but never stored
  const float* wp = W + (size_t)ncl * K + khalf * 32;
  const unsigned ldsBase =
      (unsigned)(uintptr_t)(as3_float*)(&Bsf[0][0]) +
      (unsigned)((nload * 68 + khalf * 32) * sizeof(float));

  v8f acc[4];
  #pragma unroll
  for (int nt = 0; nt < 4; ++nt)
    #pragma unroll
    for (int j = 0; j < 8; ++j) acc[nt][j] = 0.f;

  for (int k0 = 0; k0 < K; k0 += 64) {
    // ---- kick off async DMA of the 64x64 weight tile into LDS ----
    #pragma unroll
    for (int q = 0; q < 8; ++q) {
      async_b128_to_lds(ldsBase + (unsigned)(16 * q),
                        (unsigned long long)(uintptr_t)(wp + k0 + 4 * q));
    }

    // ---- build two A fragments while the DMA is in flight ----
    v16us af[2];
    #pragma unroll
    for (int kt = 0; kt < 2; ++kt) {
      v8u au;
      #pragma unroll
      for (int v = 0; v < 8; ++v) {
        const int kk = k0 + kt * 32 + kpair_base(v, half);
        if (AMODE == 0) {
          const float2 p = *(const float2*)(A1 + (size_t)row * K + kk);
          au[v] = pack_bf16x2(p.x, p.y);
        } else {
          float xv[2];
          #pragma unroll
          for (int e = 0; e < 2; ++e) {
            const int kke = kk + e;
            const int cin = kke / KW;            // compile-time KW
            const int kw  = kke - cin * KW;
            const int ls  = cl + kw - pad;
            const bool ok = (unsigned)ls < (unsigned)Lin;
            const int lsc = ok ? ls : 0;         // clamped, branchless
            const float* base = (cin < C1)
                ? (A1 + ((size_t)cb * C1 + cin) * Lin)
                : (A2 + ((size_t)cb * C2 + (cin - C1)) * Lin);
            const float t = base[lsc];
            xv[e] = ok ? t : 0.f;
          }
          au[v] = pack_bf16x2(xv[0], xv[1]);
        }
      }
      af[kt] = __builtin_bit_cast(v16us, au);
    }

    // ---- own-wave wait, then convert the region this thread DMA'd ----
    wait_async0();
    #pragma unroll
    for (int q = 0; q < 16; ++q) {
      const float2 p = *(const float2*)&Bsf[nload][khalf * 32 + 2 * q];
      *(unsigned*)&Bs16[nload][khalf * 32 + 2 * q] = pack_bf16x2(p.x, p.y);
    }
    __syncthreads();   // converted tile visible block-wide

    // ---- per K-tile: batch-load 4 B fragments from LDS, then 4 WMMAs ----
    #pragma unroll
    for (int kt = 0; kt < 2; ++kt) {
      v8u bu[4];
      #pragma unroll
      for (int nt = 0; nt < 4; ++nt)
        #pragma unroll
        for (int v = 0; v < 8; ++v) {
          const int kb = kt * 32 + kpair_base(v, half);
          bu[nt][v] = *(const unsigned*)&Bs16[nt * 16 + lm][kb];
        }
      #pragma unroll
      for (int nt = 0; nt < 4; ++nt)
        acc[nt] = wmma_bf16(af[kt], __builtin_bit_cast(v16us, bu[nt]),
                            acc[nt]);
    }
    __syncthreads();   // all waves done with Bs16/Bsf before next stage
  }

  // ---- store: C/D layout: VGPR j, lanes0-15 -> M=j, lanes16-31 -> M=j+8 ----
  #pragma unroll
  for (int nt = 0; nt < 4; ++nt) {
    const int ncol = tileN0 + nt * 16 + lm;
    if (ncol >= N) continue;
    const float bv = bias ? bias[ncol] : 0.f;
    #pragma unroll
    for (int j = 0; j < 8; ++j) {
      const int rm = tileM + half * 8 + j;
      if (rm >= M) continue;
      const float val = acc[nt][j] + bv;
      if (CMODE == 0) {
        Cmat[(size_t)rm * N + ncol] = val;
      } else {
        const int bb = rm >> lSh, ll = rm - (bb << lSh);
        Cmat[(((size_t)bb * N + ncol) << lSh) + ll] = val;
      }
    }
  }
}

// ---------------------------------------------------------------------------
// Causal depthwise conv (d_conv=4, left pad 3) + SiLU.  xi = xz[:, :di]
// xz: (M, 2*di) row-major, out: (M, di).  di = 1<<diSh, L = 1<<lSh.
// ---------------------------------------------------------------------------
__global__ __launch_bounds__(256) void dwconv_silu_k(
    const float* __restrict__ xz, const float* __restrict__ cw,
    const float* __restrict__ cb, float* __restrict__ out,
    int total, int lSh, int diSh)
{
  const int idx = blockIdx.x * blockDim.x + threadIdx.x;
  if (idx >= total) return;
  const int di = 1 << diSh;
  const int d = idx & (di - 1);
  const int m = idx >> diSh;
  const int l = m & ((1 << lSh) - 1);
  const int b = m >> lSh;
  float acc = cb[d];
  #pragma unroll
  for (int k = 0; k < 4; ++k) {
    const int ls  = l + k - 3;
    const float wv = (ls >= 0) ? cw[d * 4 + k] : 0.f;   // branchless pad
    const int lsc = (ls >= 0) ? ls : 0;
    acc += wv * xz[((size_t)(((b << lSh) + lsc)) << (diSh + 1)) + d];
  }
  out[((size_t)m << diSh) + d] = acc / (1.f + __expf(-acc));   // SiLU
}

// ---------------------------------------------------------------------------
// dt = softplus(proj[:, :r] @ dt_w^T + dt_b)   (K = r <= 16: scalar kernel)
// ---------------------------------------------------------------------------
__global__ __launch_bounds__(256) void dt_softplus_k(
    const float* __restrict__ proj, const float* __restrict__ dtw,
    const float* __restrict__ dtb, float* __restrict__ dt,
    int total, int diSh, int r, int pw)
{
  const int idx = blockIdx.x * blockDim.x + threadIdx.x;
  if (idx >= total) return;
  const int d = idx & ((1 << diSh) - 1);
  const int m = idx >> diSh;
  float s = dtb[d];
  for (int j = 0; j < r; ++j)
    s += proj[(size_t)m * pw + j] * dtw[d * r + j];
  dt[((size_t)m << diSh) + d] = (s > 20.f) ? s : log1pf(__expf(s));
}

// ---------------------------------------------------------------------------
// Fused selective scan + SiLU gate.
// One thread per (b, d-channel); 16 SSM states live in VGPRs; serial over L.
// ---------------------------------------------------------------------------
__global__ __launch_bounds__(256) void scan_gate_k(
    const float* __restrict__ xia, const float* __restrict__ dt,
    const float* __restrict__ proj, const float* __restrict__ Alog,
    const float* __restrict__ Dp, const float* __restrict__ xz,
    float* __restrict__ y, int nThreads, int L, int lSh, int diSh,
    int r, int pw)
{
  const int idx = blockIdx.x * blockDim.x + threadIdx.x;
  if (idx >= nThreads) return;
  const int di = 1 << diSh;
  const int d = idx & (di - 1);
  const int b = idx >> diSh;

  float Ar[16];
  #pragma unroll
  for (int n = 0; n < 16; ++n) Ar[n] = -__expf(Alog[d * 16 + n]);
  const float Dv = Dp[d];

  float h[16];
  #pragma unroll
  for (int n = 0; n < 16; ++n) h[n] = 0.f;

  for (int l = 0; l < L; ++l) {
    const int m = (b << lSh) + l;
    const float u   = xia[((size_t)m << diSh) + d];
    const float dtv = dt[((size_t)m << diSh) + d];
    const float* pr = proj + (size_t)m * pw;
    if (l + 1 < L) {   // hint the serial-dependency loads for the next step
      __builtin_prefetch(xia + ((size_t)(m + 1) << diSh) + d, 0, 1);
      __builtin_prefetch(dt  + ((size_t)(m + 1) << diSh) + d, 0, 1);
    }
    float acc = 0.f;
    #pragma unroll
    for (int n = 0; n < 16; ++n) {
      const float dA = __expf(dtv * Ar[n]);
      h[n] = dA * h[n] + (dtv * u) * pr[r + n];
      acc += h[n] * pr[r + 16 + n];
    }
    const float z  = xz[((size_t)m << (diSh + 1)) + di + d];
    const float sz = z / (1.f + __expf(-z));
    y[((size_t)m << diSh) + d] = (acc + u * Dv) * sz;
  }
}

// ---------------------------------------------------------------------------
// LayerNorm over channels + residual add; residual/out in (B,C,L) layout.
// xin: (M=B*L, d) row-major (GEMM output). One wave32 per row.
// ---------------------------------------------------------------------------
__global__ __launch_bounds__(256) void ln_residual_k(
    const float* __restrict__ xin, const float* __restrict__ res_bcl,
    const float* __restrict__ g,   const float* __restrict__ beta,
    float* __restrict__ out_bcl, int M, int d, int lSh)
{
  const int row  = blockIdx.x * 8 + (threadIdx.x >> 5);
  const int lane = threadIdx.x & 31;
  if (row >= M) return;
  const float* xr = xin + (size_t)row * d;
  const int b = row >> lSh;
  const int l = row & ((1 << lSh) - 1);

  float s = 0.f;
  for (int i = lane; i < d; i += 32) s += xr[i];
  #pragma unroll
  for (int off = 16; off; off >>= 1) s += __shfl_xor(s, off, 32);
  const float mean = s / (float)d;

  float v = 0.f;
  for (int i = lane; i < d; i += 32) { const float t = xr[i] - mean; v += t * t; }
  #pragma unroll
  for (int off = 16; off; off >>= 1) v += __shfl_xor(v, off, 32);
  const float inv = rsqrtf(v / (float)d + 1e-5f);

  for (int i = lane; i < d; i += 32) {
    const size_t o = (((size_t)b * d + i) << lSh) + l;
    out_bcl[o] = (xr[i] - mean) * inv * g[i] + beta[i] + res_bcl[o];
  }
}

// ---------------------------------------------------------------------------
// maxpool2 over last dim: (B*C, 2*Lo) -> (B*C, Lo), Lo = 1<<loSh
// ---------------------------------------------------------------------------
__global__ __launch_bounds__(256) void maxpool2_k(
    const float* __restrict__ in, float* __restrict__ out, int total, int loSh)
{
  const int idx = blockIdx.x * blockDim.x + threadIdx.x;
  if (idx >= total) return;
  const int l  = idx & ((1 << loSh) - 1);
  const int bc = idx >> loSh;
  const float* p = in + ((size_t)bc << (loSh + 1)) + 2 * l;
  out[idx] = fmaxf(p[0], p[1]);
}

// ---------------------------------------------------------------------------
// linear interpolation x2 (align_corners=False): (B*C, Lin) -> (B*C, Lout)
// ---------------------------------------------------------------------------
__global__ __launch_bounds__(256) void interp_k(
    const float* __restrict__ in, float* __restrict__ out,
    int total, int Lin, int loutSh)
{
  const int idx = blockIdx.x * blockDim.x + threadIdx.x;
  if (idx >= total) return;
  const int Lout = 1 << loutSh;
  const int l  = idx & (Lout - 1);
  const int bc = idx >> loutSh;
  const float scale = (float)Lin / (float)Lout;
  float src = ((float)l + 0.5f) * scale - 0.5f;
  src = fminf(fmaxf(src, 0.f), (float)(Lin - 1));
  const int i0 = (int)floorf(src);
  const int i1 = min(i0 + 1, Lin - 1);
  const float w = src - (float)i0;
  const float* p = in + (size_t)bc * Lin;
  out[idx] = p[i0] * (1.f - w) + p[i1] * w;
}

// ===========================================================================
// Host-side driver
// ===========================================================================
namespace {

inline int ilog2i(int x) { int s = 0; while ((1 << s) < x) ++s; return s; }

// A dense row-major, C row-major
inline void gemm_dense(hipStream_t s, const float* A, const float* W,
                       const float* bias, float* C, int M, int N, int K) {
  dim3 g((N + 63) / 64, (M + 63) / 64);
  wmma_gemm_t<0, 1, 0><<<g, 128, 0, s>>>(A, nullptr, W, bias, C, M, N, K,
                                         0, 0, 0, 0, 0);
}

// A channel-major (B,Cin,Lc) pointwise; C row-major (B*Lc, N)
inline void gemm_pw_row(hipStream_t s, const float* A, const float* W,
                        const float* bias, float* C, int Bn, int Lc,
                        int N, int Cin) {
  const int M = Bn * Lc;
  dim3 g((N + 63) / 64, (M + 63) / 64);
  wmma_gemm_t<1, 1, 0><<<g, 128, 0, s>>>(A, nullptr, W, bias, C, M, N, Cin,
                                         Cin, 0, Lc, ilog2i(Lc), 0);
}

// A channel-major (B,Cin,Lc) pointwise; C channel-major (B,N,Lc)
inline void gemm_pw_bcl(hipStream_t s, const float* A, const float* W,
                        const float* bias, float* C, int Bn, int Lc,
                        int N, int Cin) {
  const int M = Bn * Lc;
  dim3 g((N + 63) / 64, (M + 63) / 64);
  wmma_gemm_t<1, 1, 1><<<g, 128, 0, s>>>(A, nullptr, W, bias, C, M, N, Cin,
                                         Cin, 0, Lc, ilog2i(Lc), 0);
}

// K=3 same-pad conv over concat(A1:C1, A2:C2); C channel-major (B,N,Lc)
inline void gemm_conv3(hipStream_t s, const float* A1, const float* A2,
                       const float* W, const float* bias, float* C,
                       int Bn, int Lc, int N, int C1, int C2) {
  const int M = Bn * Lc;
  const int K = (C1 + C2) * 3;
  dim3 g((N + 63) / 64, (M + 63) / 64);
  wmma_gemm_t<1, 3, 1><<<g, 128, 0, s>>>(A1, A2, W, bias, C, M, N, K,
                                         C1, C2, Lc, ilog2i(Lc), 1);
}

struct MambaP {
  const float *in_w, *cw, *cb, *xw, *dtw, *dtb, *Alog, *Dp, *ow, *g, *beta;
  int d, di, r, pw;
};

inline MambaP mk_mamba(void* const* in, int base, int d) {
  MambaP p;
  p.in_w = (const float*)in[base + 0];
  p.cw   = (const float*)in[base + 1];
  p.cb   = (const float*)in[base + 2];
  p.xw   = (const float*)in[base + 3];
  p.dtw  = (const float*)in[base + 4];
  p.dtb  = (const float*)in[base + 5];
  p.Alog = (const float*)in[base + 6];
  p.Dp   = (const float*)in[base + 7];
  p.ow   = (const float*)in[base + 8];
  p.g    = (const float*)in[base + 9];
  p.beta = (const float*)in[base + 10];
  p.d = d; p.di = 2 * d; p.r = (d + 15) / 16; p.pw = p.r + 32;
  return p;
}

struct Scratch { float *XZ, *XIA, *PROJ, *DT, *Y, *MO; };

// xin/xout are channel-major (B, d, L)
inline void ssm_block(hipStream_t s, const MambaP& p, const float* xin,
                      float* xout, int Bn, int L, const Scratch& w) {
  const int M = Bn * L;
  const int lSh = ilog2i(L), diSh = ilog2i(p.di);
  // in_proj (reads BCL directly) -> xz (M, 2di) row-major
  gemm_pw_row(s, xin, p.in_w, nullptr, w.XZ, Bn, L, 2 * p.di, p.d);
  // causal depthwise conv + silu -> xia (M, di)
  { int t = M * p.di;
    dwconv_silu_k<<<(t + 255) / 256, 256, 0, s>>>(w.XZ, p.cw, p.cb, w.XIA,
                                                  t, lSh, diSh); }
  // x_proj -> proj (M, r+2n)
  gemm_dense(s, w.XIA, p.xw, nullptr, w.PROJ, M, p.pw, p.di);
  // dt = softplus(dt_proj)
  { int t = M * p.di;
    dt_softplus_k<<<(t + 255) / 256, 256, 0, s>>>(w.PROJ, p.dtw, p.dtb, w.DT,
                                                  t, diSh, p.r, p.pw); }
  // selective scan + gate -> y (M, di)
  { int t = Bn * p.di;
    scan_gate_k<<<(t + 255) / 256, 256, 0, s>>>(w.XIA, w.DT, w.PROJ, p.Alog,
                                                p.Dp, w.XZ, w.Y,
                                                t, L, lSh, diSh, p.r, p.pw); }
  // out_proj -> mo (M, d) row-major
  gemm_dense(s, w.Y, p.ow, nullptr, w.MO, M, p.d, p.di);
  // LayerNorm + residual, writes channel-major
  ln_residual_k<<<(M + 7) / 8, 256, 0, s>>>(w.MO, xin, p.g, p.beta, xout,
                                            M, p.d, lSh);
}

// workspace layout (floats)
constexpr size_t OFF_XZ   = 0;
constexpr size_t OFF_XIA  = OFF_XZ   + 16777216;
constexpr size_t OFF_PROJ = OFF_XIA  + 8388608;
constexpr size_t OFF_DT   = OFF_PROJ + 4194304;
constexpr size_t OFF_Y    = OFF_DT   + 8388608;
constexpr size_t OFF_MO   = OFF_Y    + 8388608;
constexpr size_t OFF_BIN  = OFF_MO   + 4194304;
constexpr size_t OFF_BOUT = OFF_BIN  + 4194304;
constexpr size_t OFF_X1   = OFF_BOUT + 4194304;
constexpr size_t OFF_X2   = OFF_X1   + 4194304;
constexpr size_t OFF_X3   = OFF_X2   + 4194304;
constexpr size_t OFF_TMPA = OFF_X3   + 4194304;
constexpr size_t OFF_TMPB = OFF_TMPA + 4194304;

} // namespace

extern "C" void kernel_launch(void* const* d_in, const int* in_sizes, int n_in,
                              void* d_out, int out_size, void* d_ws,
                              size_t ws_size, hipStream_t stream) {
  (void)in_sizes; (void)n_in; (void)out_size; (void)ws_size;
  const int Bn = 16, L = 4096;
  float* ws = (float*)d_ws;
  float* XZ   = ws + OFF_XZ;
  float* XIA  = ws + OFF_XIA;
  float* PROJ = ws + OFF_PROJ;
  float* DT   = ws + OFF_DT;
  float* Y    = ws + OFF_Y;
  float* MO   = ws + OFF_MO;
  float* BIN  = ws + OFF_BIN;
  float* BOUT = ws + OFF_BOUT;
  float* X1   = ws + OFF_X1;
  float* X2   = ws + OFF_X2;
  float* X3   = ws + OFF_X3;
  float* TMPA = ws + OFF_TMPA;
  float* TMPB = ws + OFF_TMPB;
  Scratch sc{XZ, XIA, PROJ, DT, Y, MO};

  const float* x = (const float*)d_in[0];
  #define IN(i) ((const float*)d_in[(i)])

  // ---- embedding conv (B,64,L) + enc1 (all tensors stay channel-major) ----
  gemm_conv3(stream, x, nullptr, IN(1), IN(2), TMPA, Bn, L, 64, 64, 0);
  gemm_pw_bcl(stream, TMPA, IN(3), IN(4), BIN, Bn, L, 64, 64);
  ssm_block(stream, mk_mamba(d_in, 5, 64), BIN, X1, Bn, L, sc);

  // ---- encoder 2/3 ----
  maxpool2_k<<<(Bn * 64 * 2048 + 255) / 256, 256, 0, stream>>>(
      X1, TMPA, Bn * 64 * 2048, 11);
  gemm_conv3(stream, TMPA, nullptr, IN(16), IN(17), X2, Bn, 2048, 128, 64, 0);
  maxpool2_k<<<(Bn * 128 * 1024 + 255) / 256, 256, 0, stream>>>(
      X2, TMPA, Bn * 128 * 1024, 10);
  gemm_conv3(stream, TMPA, nullptr, IN(18), IN(19), X3, Bn, 1024, 256, 128, 0);

  // ---- bottleneck ----
  maxpool2_k<<<(Bn * 256 * 512 + 255) / 256, 256, 0, stream>>>(
      X3, TMPA, Bn * 256 * 512, 9);
  gemm_pw_bcl(stream, TMPA, IN(20), IN(21), BIN, Bn, 512, 256, 256);
  ssm_block(stream, mk_mamba(d_in, 22, 256), BIN, BOUT, Bn, 512, sc);
  gemm_pw_bcl(stream, BOUT, IN(33), IN(34), TMPA, Bn, 512, 256, 256);
  interp_k<<<(Bn * 256 * 1024 + 255) / 256, 256, 0, stream>>>(
      TMPA, TMPB, Bn * 256 * 1024, 512, 10);

  // ---- decoder 3 ----
  gemm_conv3(stream, X3, TMPB, IN(35), IN(36), TMPA, Bn, 1024, 256, 256, 256);
  ssm_block(stream, mk_mamba(d_in, 37, 256), TMPA, BOUT, Bn, 1024, sc);
  interp_k<<<(Bn * 256 * 2048 + 255) / 256, 256, 0, stream>>>(
      BOUT, TMPB, Bn * 256 * 2048, 1024, 11);

  // ---- decoder 2 ----
  gemm_conv3(stream, X2, TMPB, IN(48), IN(49), TMPA, Bn, 2048, 128, 128, 256);
  ssm_block(stream, mk_mamba(d_in, 50, 128), TMPA, BOUT, Bn, 2048, sc);
  interp_k<<<(Bn * 128 * 4096 + 255) / 256, 256, 0, stream>>>(
      BOUT, TMPB, Bn * 128 * 4096, 2048, 12);

  // ---- decoder 1 + output head ----
  gemm_conv3(stream, X1, TMPB, IN(61), IN(62), TMPA, Bn, 4096, 64, 64, 128);
  gemm_pw_bcl(stream, TMPA, IN(63), IN(64), (float*)d_out, Bn, 4096, 64, 64);
  #undef IN
}
